// SparseTransformerLayer_61392262529129
// MI455X (gfx1250) — matrix operand order repaired and checked by reading, not compiled, not run on hardware
//
#include <hip/hip_runtime.h>
#include <hip/hip_bf16.h>

typedef __attribute__((ext_vector_type(16))) _Float16 v16h;
typedef __attribute__((ext_vector_type(8)))  float    v8f;
typedef __attribute__((ext_vector_type(4)))  int      i4;

#define WMMA_F16(a, b, c) \
  __builtin_amdgcn_wmma_f32_16x16x32_f16(false, (a), false, (b), (short)0, (c), false, false)

#if defined(__HIP_DEVICE_COMPILE__) && __has_builtin(__builtin_amdgcn_global_load_async_to_lds_b128)
#define HAVE_ASYNC_LDS 1
#else
#define HAVE_ASYNC_LDS 0
#endif

// 16-byte global -> LDS copy; async (ASYNCcnt) path when the toolchain has it.
__device__ __forceinline__ void copy16B_g2l(const void* g, void* l) {
#if HAVE_ASYNC_LDS
  void* gv = const_cast<void*>(g);
  __builtin_amdgcn_global_load_async_to_lds_b128(
      (__attribute__((address_space(1))) i4*)gv,
      (__attribute__((address_space(3))) i4*)l, 0, 0);
#else
  *(uint4*)l = *(const uint4*)g;
#endif
}

__device__ __forceinline__ void wait_async_copies() {
#if HAVE_ASYNC_LDS
#if __has_builtin(__builtin_amdgcn_s_wait_asynccnt)
  __builtin_amdgcn_s_wait_asynccnt(0);
#else
  asm volatile("s_wait_asynccnt 0x0" ::: "memory");
#endif
#endif
}

// ---------------------------------------------------------------- constants
namespace cfg {
constexpr int B  = 2;
constexpr int S  = 2048;
constexpr int D  = 1024;
constexpr int F  = 4096;
constexpr int H  = 16;
constexpr int Dh = 64;
constexpr int M  = B * S;          // 4096 rows of activations
constexpr int HALFW = 204;         // w = int(2048*0.2)=409, w//2 = 204
}

// ---------------------------------------------------------------- f32 -> f16
__global__ void cvt_f32_f16(const float* __restrict__ in,
                            _Float16* __restrict__ out, int n) {
  int i = blockIdx.x * blockDim.x + threadIdx.x;
  if (i < n) out[i] = (_Float16)in[i];
}

// ---------------------------------------------------------------- LayerNorm (f32 in, f16 out)
__global__ __launch_bounds__(256) void ln_kernel(const float* __restrict__ x,
                                                 const float* __restrict__ gamma,
                                                 const float* __restrict__ beta,
                                                 _Float16* __restrict__ out) {
  __shared__ float red[256];
  const int row = blockIdx.x;
  const int t   = threadIdx.x;
  const float* xr = x + (size_t)row * cfg::D;
  float4 v = ((const float4*)xr)[t];                 // 256 threads * 4 = 1024
  float s = v.x + v.y + v.z + v.w;
  red[t] = s; __syncthreads();
  for (int off = 128; off > 0; off >>= 1) {
    if (t < off) red[t] += red[t + off];
    __syncthreads();
  }
  float mean = red[0] / cfg::D;
  __syncthreads();
  float dx = v.x - mean, dy = v.y - mean, dz = v.z - mean, dw = v.w - mean;
  red[t] = dx*dx + dy*dy + dz*dz + dw*dw; __syncthreads();
  for (int off = 128; off > 0; off >>= 1) {
    if (t < off) red[t] += red[t + off];
    __syncthreads();
  }
  float rstd = rsqrtf(red[0] / cfg::D + 1e-5f);
  int c = t * 4;
  _Float16* o = out + (size_t)row * cfg::D + c;
  o[0] = (_Float16)(dx * rstd * gamma[c + 0] + beta[c + 0]);
  o[1] = (_Float16)(dy * rstd * gamma[c + 1] + beta[c + 1]);
  o[2] = (_Float16)(dz * rstd * gamma[c + 2] + beta[c + 2]);
  o[3] = (_Float16)(dw * rstd * gamma[c + 3] + beta[c + 3]);
}

// ---------------------------------------------------------------- generic WMMA GEMM
// C[M,N] = A[M,K] (f16, row major) * Bw[K,N] (f16, row major) + bias
// modes: 0 = f16 row-major out; 1 = f16 [B,H,S,Dh] out (Q,K);
//        2 = f16 [B,H,Dh,S] out (V transposed); 3 = f32 out + f32 residual
#define GT_M 128
#define GT_N 128
#define GT_K 32
__global__ __launch_bounds__(256) void gemm_f16(const _Float16* __restrict__ A,
                                                const _Float16* __restrict__ Bw,
                                                const float* __restrict__ bias,
                                                const float* __restrict__ resid,
                                                _Float16* __restrict__ outH,
                                                float* __restrict__ outF,
                                                int Mdim, int Ndim, int Kdim,
                                                int mode, int do_relu) {
  __shared__ _Float16 lsA[GT_M][GT_K];       // [m][k]
  __shared__ _Float16 lsB[GT_N][GT_K];       // transposed: [n][k]
  const int tid  = threadIdx.x;
  const int lane = tid & 31;
  const int wave = tid >> 5;
  const int g    = lane >> 4;                // lane half (0/1)
  const int ln   = lane & 15;
  const int wm   = (wave & 3) * 32;          // wave row offset inside tile
  const int wn   = (wave >> 2) * 64;         // wave col offset inside tile
  const int bm   = blockIdx.x * GT_M;
  const int bn   = blockIdx.y * GT_N;

  v8f acc[2][4] = {};

  for (int k0 = 0; k0 < Kdim; k0 += GT_K) {
    // stage A tile (contiguous): async global->LDS, 2 x 16B per thread
    {
      int r = tid >> 1, c = (tid & 1) * 16;
      const _Float16* src = A + (size_t)(bm + r) * Kdim + k0 + c;
      _Float16* dst = &lsA[r][c];
      copy16B_g2l(src, dst);
      copy16B_g2l(src + 8, dst + 8);
      if (k0 + GT_K < Kdim)
        __builtin_prefetch((const void*)(src + GT_K), 0, 0);
    }
    // stage B tile transposed: thread -> k row tid/8, 16 n values at (tid&7)*16
    {
      int kk = tid >> 3;
      int nn = (tid & 7) * 16;
      const _Float16* src = Bw + (size_t)(k0 + kk) * Ndim + bn + nn;
#pragma unroll
      for (int e = 0; e < 16; ++e) lsB[nn + e][kk] = src[e];
    }
    wait_async_copies();
    __syncthreads();

    // A fragments: lane holds row m, half h -> k = 16g+h (contiguous)
    v16h af[2];
#pragma unroll
    for (int i = 0; i < 2; ++i) {
      const _Float16* row = &lsA[wm + i * 16 + ln][0];
#pragma unroll
      for (int h = 0; h < 8; ++h) {
        af[i][h]     = row[16 * g + h];
        af[i][8 + h] = row[16 * g + 8 + h];
      }
    }
    // B fragments: lane holds col n, half h -> k = 16g+h  (contiguous in lsB[n][k])
    v16h bf[4];
#pragma unroll
    for (int j = 0; j < 4; ++j) {
      const _Float16* col = &lsB[wn + j * 16 + ln][0];
#pragma unroll
      for (int h = 0; h < 16; ++h) bf[j][h] = col[16 * g + h];
    }
#pragma unroll
    for (int i = 0; i < 2; ++i)
#pragma unroll
      for (int j = 0; j < 4; ++j)
        acc[i][j] = WMMA_F16(af[i], bf[j], acc[i][j]);
    __syncthreads();
  }

  // epilogue: C layout -> vgpr r, lane: M = r + 8g, N = ln
#pragma unroll
  for (int i = 0; i < 2; ++i) {
#pragma unroll
    for (int j = 0; j < 4; ++j) {
#pragma unroll
      for (int r = 0; r < 8; ++r) {
        int m = bm + wm + i * 16 + r + 8 * g;
        int n = bn + wn + j * 16 + ln;
        float val = acc[i][j][r] + bias[n];
        if (do_relu) val = fmaxf(val, 0.0f);
        if (mode == 0) {
          outH[(size_t)m * Ndim + n] = (_Float16)val;
        } else if (mode == 1) {          // [B,H,S,Dh]
          int b = m >> 11, s = m & (cfg::S - 1);
          int h = n >> 6,  dh = n & (cfg::Dh - 1);
          outH[(((size_t)(b * cfg::H + h) * cfg::S) + s) * cfg::Dh + dh] = (_Float16)val;
        } else if (mode == 2) {          // [B,H,Dh,S]
          int b = m >> 11, s = m & (cfg::S - 1);
          int h = n >> 6,  dh = n & (cfg::Dh - 1);
          outH[(((size_t)(b * cfg::H + h) * cfg::Dh) + dh) * cfg::S + s] = (_Float16)val;
        } else {                         // f32 + residual
          size_t idx = (size_t)m * Ndim + n;
          outF[idx] = val + resid[idx];
        }
      }
    }
  }
}

// ---------------------------------------------------------------- local-window flash attention
// Q,K: [B,H,S,Dh] f16 ; Vt: [B,H,Dh,S] f16 ; Out: [B,S,H*Dh] f16
__global__ __launch_bounds__(256) void attn_kernel(const _Float16* __restrict__ Q,
                                                   const _Float16* __restrict__ Kh,
                                                   const _Float16* __restrict__ Vt,
                                                   _Float16* __restrict__ Out) {
  using namespace cfg;
  __shared__ _Float16 lp[8][16][32];           // per-wave P tile
  const int tid  = threadIdx.x;
  const int lane = tid & 31;
  const int wave = tid >> 5;
  const int g    = lane >> 4;
  const int ln   = lane & 15;
  const int gw   = blockIdx.x * 8 + wave;      // global wave = query tile id
  const int qt   = gw & 127;                   // S/16 tiles
  const int bh   = gw >> 7;                    // b*H + h
  const int b    = bh >> 4;
  const int h    = bh & 15;
  const int q0   = qt * 16;
  const size_t base = (size_t)bh * S * Dh;
  const float scale = 0.125f;                  // 1/sqrt(64)

  // Q fragments (K-dim 64 = two 16x32 A fragments), contiguous global loads
  v16h aq0, aq1;
  {
    const _Float16* qrow = Q + base + (size_t)(q0 + ln) * Dh;
#pragma unroll
    for (int hh = 0; hh < 8; ++hh) {
      aq0[hh]     = qrow[16 * g + hh];
      aq0[8 + hh] = qrow[16 * g + 8 + hh];
      aq1[hh]     = qrow[32 + 16 * g + hh];
      aq1[8 + hh] = qrow[40 + 16 * g + hh];
    }
  }

  v8f oacc[4] = {};
  float mrun[8], lrun[8];
#pragma unroll
  for (int r = 0; r < 8; ++r) { mrun[r] = -1e30f; lrun[r] = 0.0f; }

  int jmin = q0 - HALFW;        if (jmin < 0) jmin = 0;
  int jmax = q0 + 15 + HALFW;   if (jmax > S - 1) jmax = S - 1;

  for (int kb = jmin & ~31; kb <= jmax; kb += 32) {
    // ----- scores: S[16q x 32k] via two 16-key chunks, 2 WMMAs each
    v8f sc[2];
#pragma unroll
    for (int c = 0; c < 2; ++c) {
      int key = kb + 16 * c + ln;
      if (key > S - 1) key = S - 1;            // clamp load; masked below
      const _Float16* krow = Kh + base + (size_t)key * Dh;
      v16h bk0, bk1;
#pragma unroll
      for (int hh = 0; hh < 16; ++hh) {
        bk0[hh] = krow[16 * g + hh];
        bk1[hh] = krow[32 + 16 * g + hh];
      }
      v8f z = {};
      z = WMMA_F16(aq0, bk0, z);
      z = WMMA_F16(aq1, bk1, z);
      sc[c] = z;
    }
    // ----- online softmax, rows M = r + 8g live across lanes of each 16-lane half
#pragma unroll
    for (int r = 0; r < 8; ++r) {
      int i  = q0 + r + 8 * g;
      int j0 = kb + ln, j1 = kb + 16 + ln;
      bool v0 = (j0 >= i - HALFW) && (j0 <= i + HALFW) && (j0 < S);
      bool v1 = (j1 >= i - HALFW) && (j1 <= i + HALFW) && (j1 < S);
      float s0 = v0 ? sc[0][r] * scale : -1e30f;
      float s1 = v1 ? sc[1][r] * scale : -1e30f;
      float mc = fmaxf(s0, s1);
#pragma unroll
      for (int off = 1; off < 16; off <<= 1) mc = fmaxf(mc, __shfl_xor(mc, off, 16));
      float mnew  = fmaxf(mrun[r], mc);
      float p0    = v0 ? __expf(s0 - mnew) : 0.0f;
      float p1    = v1 ? __expf(s1 - mnew) : 0.0f;
      float alpha = __expf(mrun[r] - mnew);
      float ladd  = p0 + p1;
#pragma unroll
      for (int off = 1; off < 16; off <<= 1) ladd += __shfl_xor(ladd, off, 16);
      lrun[r] = lrun[r] * alpha + ladd;
      mrun[r] = mnew;
#pragma unroll
      for (int j = 0; j < 4; ++j) oacc[j][r] *= alpha;
      lp[wave][r + 8 * g][ln]      = (_Float16)p0;
      lp[wave][r + 8 * g][16 + ln] = (_Float16)p1;
    }
    // ----- O += P[16x32] * V[32x64]  (P re-laid-out via LDS into A fragment)
    v16h ap;
    {
      const _Float16* prow = &lp[wave][ln][0];
#pragma unroll
      for (int hh = 0; hh < 8; ++hh) {
        ap[hh]     = prow[16 * g + hh];
        ap[8 + hh] = prow[16 * g + 8 + hh];
      }
    }
#pragma unroll
    for (int j = 0; j < 4; ++j) {
      const _Float16* vcol = Vt + ((size_t)bh * Dh + 16 * j + ln) * S;
      v16h bv;
#pragma unroll
      for (int hh = 0; hh < 16; ++hh) bv[hh] = vcol[kb + 16 * g + hh];
      oacc[j] = WMMA_F16(ap, bv, oacc[j]);
    }
  }

  // ----- normalize + write [B,S,H*Dh]
#pragma unroll
  for (int r = 0; r < 8; ++r) {
    float inv = 1.0f / lrun[r];
    int srow = q0 + r + 8 * g;
#pragma unroll
    for (int j = 0; j < 4; ++j)
      Out[((size_t)b * S + srow) * D + h * Dh + 16 * j + ln] =
          (_Float16)(oacc[j][r] * inv);
  }
}

// ---------------------------------------------------------------- launcher
extern "C" void kernel_launch(void* const* d_in, const int* in_sizes, int n_in,
                              void* d_out, int out_size, void* d_ws, size_t ws_size,
                              hipStream_t stream) {
  using namespace cfg;
  const float* x     = (const float*)d_in[0];
  const float* wq    = (const float*)d_in[1];
  const float* bq    = (const float*)d_in[2];
  const float* wk    = (const float*)d_in[3];
  const float* bk    = (const float*)d_in[4];
  const float* wv    = (const float*)d_in[5];
  const float* bv    = (const float*)d_in[6];
  const float* wo    = (const float*)d_in[7];
  const float* bo    = (const float*)d_in[8];
  const float* ln1g  = (const float*)d_in[9];
  const float* ln1b  = (const float*)d_in[10];
  const float* ln2g  = (const float*)d_in[11];
  const float* ln2b  = (const float*)d_in[12];
  const float* w1    = (const float*)d_in[13];
  const float* b1    = (const float*)d_in[14];
  const float* w2    = (const float*)d_in[15];
  const float* b2    = (const float*)d_in[16];
  float* out = (float*)d_out;

  size_t off = 0;
  auto take = [&](size_t bytes) {
    void* p = (char*)d_ws + off;
    off += (bytes + 255) & ~(size_t)255;
    return p;
  };
  _Float16* wq_h   = (_Float16*)take((size_t)D * D * 2);
  _Float16* wk_h   = (_Float16*)take((size_t)D * D * 2);
  _Float16* wv_h   = (_Float16*)take((size_t)D * D * 2);
  _Float16* wo_h   = (_Float16*)take((size_t)D * D * 2);
  _Float16* w1_h   = (_Float16*)take((size_t)D * F * 2);
  _Float16* w2_h   = (_Float16*)take((size_t)F * D * 2);
  _Float16* xn_h   = (_Float16*)take((size_t)M * D * 2);   // reused for LN2 output
  _Float16* q_h    = (_Float16*)take((size_t)M * D * 2);
  _Float16* k_h    = (_Float16*)take((size_t)M * D * 2);
  _Float16* vt_h   = (_Float16*)take((size_t)M * D * 2);
  _Float16* attn_h = (_Float16*)take((size_t)M * D * 2);
  float*    x1_f   = (float*)take((size_t)M * D * 4);
  _Float16* ff1_h  = q_h;   // alias q/k/vt/attn region (exactly M*F*2 bytes), free after attention

  // weight conversion
  auto cvt = [&](const float* src, _Float16* dst, int n) {
    cvt_f32_f16<<<(n + 255) / 256, 256, 0, stream>>>(src, dst, n);
  };
  cvt(wq, wq_h, D * D);
  cvt(wk, wk_h, D * D);
  cvt(wv, wv_h, D * D);
  cvt(wo, wo_h, D * D);
  cvt(w1, w1_h, D * F);
  cvt(w2, w2_h, F * D);

  // LN1
  ln_kernel<<<M, 256, 0, stream>>>(x, ln1g, ln1b, xn_h);

  // QKV projections (WMMA GEMMs)
  dim3 gemmBlk(256);
  dim3 gDD(M / GT_M, D / GT_N);
  gemm_f16<<<gDD, gemmBlk, 0, stream>>>(xn_h, wq_h, bq, nullptr, q_h,  nullptr, M, D, D, 1, 0);
  gemm_f16<<<gDD, gemmBlk, 0, stream>>>(xn_h, wk_h, bk, nullptr, k_h,  nullptr, M, D, D, 1, 0);
  gemm_f16<<<gDD, gemmBlk, 0, stream>>>(xn_h, wv_h, bv, nullptr, vt_h, nullptr, M, D, D, 2, 0);

  // local-window flash attention: 4096 query tiles, 8 waves/block
  attn_kernel<<<512, 256, 0, stream>>>(q_h, k_h, vt_h, attn_h);

  // output projection + residual -> x1 (f32)
  gemm_f16<<<gDD, gemmBlk, 0, stream>>>(attn_h, wo_h, bo, x, nullptr, x1_f, M, D, D, 3, 0);

  // LN2 (reuse xn_h)
  ln_kernel<<<M, 256, 0, stream>>>(x1_f, ln2g, ln2b, xn_h);

  // FFN
  dim3 gDF(M / GT_M, F / GT_N);
  gemm_f16<<<gDF, gemmBlk, 0, stream>>>(xn_h, w1_h, b1, nullptr, ff1_h, nullptr, M, F, D, 0, 1);
  gemm_f16<<<gDD, gemmBlk, 0, stream>>>(ff1_h, w2_h, b2, x1_f, nullptr, out, M, D, F, 3, 0);
}
